// Attention_16527034155395
// MI455X (gfx1250) — matrix operand order repaired and checked by reading, not compiled, not run on hardware
//
#include <hip/hip_runtime.h>

#define B_     4
#define NSEQ   2048
#define DIM_   1024
#define HEADS_ 16
#define DH_    64
#define ROWS   (B_ * NSEQ)   // 8192

typedef __attribute__((ext_vector_type(16))) __bf16 v16bf;
typedef __attribute__((ext_vector_type(8)))  float  v8f;

__device__ __forceinline__ unsigned short f2bf(float f) {
  unsigned int u = __float_as_uint(f);
  return (unsigned short)((u + 0x7FFFu + ((u >> 16) & 1u)) >> 16);  // RNE
}

// A fragment (16x32 bf16): lane holds row M = m0 + lane%16.
// elements 0..7  = K = k0 + half*8 + 0..7
// elements 8..15 = K = k0 + 16 + half*8 + 0..7
__device__ __forceinline__ v16bf load_a16(const unsigned short* A, int lda, int m0, int k0) {
  int lane = threadIdx.x & 31;
  const unsigned short* p = A + (size_t)(m0 + (lane & 15)) * lda + k0 + ((lane >> 4) << 3);
  union { uint4 q[2]; v16bf v; } f;
  f.q[0] = *(const uint4*)p;
  f.q[1] = *(const uint4*)(p + 16);
  return f.v;
}

// B fragment (32x16 bf16, K x N) from a source where output-column n is a
// row of Bm (row-major, K-contiguous): lane holds col N = n0 + lane%16,
// elements 0..7 = K = k0 + half*8 + 0..7, 8..15 = K = k0 + 16 + half*8 + 0..7.
__device__ __forceinline__ v16bf load_b16(const unsigned short* Bm, int ldb, int n0, int k0) {
  int lane = threadIdx.x & 31;
  const unsigned short* p = Bm + (size_t)(n0 + (lane & 15)) * ldb + k0 + ((lane >> 4) << 3);
  union { uint4 q[2]; v16bf v; } f;
  f.q[0] = *(const uint4*)p;
  f.q[1] = *(const uint4*)(p + 16);
  return f.v;
}

__device__ __forceinline__ v8f wmma_bf16(v16bf a, v16bf b, v8f c) {
  return __builtin_amdgcn_wmma_f32_16x16x32_bf16(false, a, false, b, (short)0, c, false, false);
}

// ---------------- weight fp32 -> bf16 ----------------
__global__ __launch_bounds__(256) void cvt_kernel(const float* __restrict__ src,
                                                  unsigned short* __restrict__ dst, int n) {
  int i = blockIdx.x * 256 + threadIdx.x;
  int stride = gridDim.x * 256;
  for (; i < n; i += stride) dst[i] = f2bf(src[i]);
}

// ---------------- LayerNorm + bf16 cast ----------------
__global__ __launch_bounds__(256) void ln_kernel(const float* __restrict__ x,
                                                 const float* __restrict__ gw,
                                                 const float* __restrict__ gb,
                                                 unsigned short* __restrict__ xn) {
  __shared__ float s1[256], s2[256];
  int row = blockIdx.x, t = threadIdx.x;
  float4 v = ((const float4*)(x + (size_t)row * DIM_))[t];
  s1[t] = v.x + v.y + v.z + v.w;
  s2[t] = v.x * v.x + v.y * v.y + v.z * v.z + v.w * v.w;
  __syncthreads();
  for (int off = 128; off > 0; off >>= 1) {
    if (t < off) { s1[t] += s1[t + off]; s2[t] += s2[t + off]; }
    __syncthreads();
  }
  float mean = s1[0] * (1.f / DIM_);
  float inv  = rsqrtf(s2[0] * (1.f / DIM_) - mean * mean + 1e-5f);
  float4 w4 = ((const float4*)gw)[t];
  float4 b4 = ((const float4*)gb)[t];
  unsigned short* o = xn + (size_t)row * DIM_ + t * 4;
  o[0] = f2bf((v.x - mean) * inv * w4.x + b4.x);
  o[1] = f2bf((v.y - mean) * inv * w4.y + b4.y);
  o[2] = f2bf((v.z - mean) * inv * w4.z + b4.z);
  o[3] = f2bf((v.w - mean) * inv * w4.w + b4.w);
}

// ---------------- QKV projection: [8192,1024] x [3072,1024]^T ----------------
// wave = one 16x64 output tile; 64 cols land inside exactly one head of q/k/v.
__global__ __launch_bounds__(256) void qkv_kernel(const unsigned short* __restrict__ xn,
                                                  const unsigned short* __restrict__ wq,
                                                  unsigned short* __restrict__ q,
                                                  unsigned short* __restrict__ k,
                                                  unsigned short* __restrict__ vt) {
  int wid = blockIdx.x * 8 + (threadIdx.x >> 5);
  int mt = wid & 511, nt = wid >> 9;          // 512 m-tiles, 48 n-tiles
  int m0 = mt << 4, e0 = nt << 6;
  v8f acc[4] = {};
  for (int k0 = 0; k0 < DIM_; k0 += 32) {
    v16bf a = load_a16(xn, DIM_, m0, k0);
#pragma unroll
    for (int t = 0; t < 4; t++)
      acc[t] = wmma_bf16(a, load_b16(wq, DIM_, e0 + t * 16, k0), acc[t]);
  }
  int lane = threadIdx.x & 31, half = lane >> 4, nl = lane & 15;
  int which = e0 >> 10;                 // 0=q, 1=k, 2=v
  int head  = (e0 & 1023) >> 6;
#pragma unroll
  for (int t = 0; t < 4; t++) {
#pragma unroll
    for (int r = 0; r < 8; r++) {
      int m = m0 + half * 8 + r;
      int bi = m >> 11, n = m & 2047;
      int d = t * 16 + nl;
      unsigned short val = f2bf(acc[t][r]);
      size_t bh = (size_t)(bi * HEADS_ + head);
      if (which == 0)      q[(bh * NSEQ + n) * DH_ + d] = val;
      else if (which == 1) k[(bh * NSEQ + n) * DH_ + d] = val;
      else                 vt[(bh * DH_ + d) * NSEQ + n] = val;   // V stored transposed
    }
  }
}

// ---------------- Flash attention: wave = 16 query rows, full j sweep ----------------
__global__ __launch_bounds__(128) void flash_kernel(const unsigned short* __restrict__ q,
                                                    const unsigned short* __restrict__ k,
                                                    const unsigned short* __restrict__ vt,
                                                    unsigned short* __restrict__ ao) {
  __shared__ __align__(16) unsigned short pst[4][16 * 32];   // per-wave P staging
  int wslot = threadIdx.x >> 5;
  int wid = blockIdx.x * 4 + wslot;
  int it = wid & 127;                 // 128 i-tiles of 16 rows
  int bh = wid >> 7;                  // 0..63 (b*H + h)
  int i0 = it << 4;
  const unsigned short* qh = q  + (size_t)bh * NSEQ * DH_;
  const unsigned short* kh = k  + (size_t)bh * NSEQ * DH_;
  const unsigned short* vh = vt + (size_t)bh * DH_ * NSEQ;

  v16bf aq0 = load_a16(qh, DH_, i0, 0);
  v16bf aq1 = load_a16(qh, DH_, i0, 32);

  float mrow[8], lrow[8];
  v8f o[4] = {};
#pragma unroll
  for (int r = 0; r < 8; r++) { mrow[r] = -1e30f; lrow[r] = 0.f; }

  int lane = threadIdx.x & 31, half = lane >> 4, nl = lane & 15;
  unsigned short* st = pst[wslot];

  for (int j0 = 0; j0 < NSEQ; j0 += 32) {
    v8f s0 = {}, s1 = {};
    s0 = wmma_bf16(aq0, load_b16(kh, DH_, j0,      0),  s0);
    s0 = wmma_bf16(aq1, load_b16(kh, DH_, j0,      32), s0);
    s1 = wmma_bf16(aq0, load_b16(kh, DH_, j0 + 16, 0),  s1);
    s1 = wmma_bf16(aq1, load_b16(kh, DH_, j0 + 16, 32), s1);

    float p0[8], p1[8];
#pragma unroll
    for (int r = 0; r < 8; r++) {
      float a = s0[r] * 0.125f, b = s1[r] * 0.125f;   // scale = 64^-0.5
      float mx = fmaxf(a, b);
#pragma unroll
      for (int off = 1; off < 16; off <<= 1) mx = fmaxf(mx, __shfl_xor(mx, off, 32));
      float nm   = fmaxf(mrow[r], mx);
      float corr = __expf(mrow[r] - nm);
      float e0v  = __expf(a - nm);
      float e1v  = __expf(b - nm);
      float rs   = e0v + e1v;
#pragma unroll
      for (int off = 1; off < 16; off <<= 1) rs += __shfl_xor(rs, off, 32);
      lrow[r] = lrow[r] * corr + rs;
      mrow[r] = nm;
      p0[r] = e0v; p1[r] = e1v;
      o[0][r] *= corr; o[1][r] *= corr; o[2][r] *= corr; o[3][r] *= corr;
    }

    // C-layout -> row-major P(16x32) in LDS (per-wave region; DS is in-order per wave)
#pragma unroll
    for (int r = 0; r < 8; r++) {
      st[(half * 8 + r) * 32 + nl]      = f2bf(p0[r]);
      st[(half * 8 + r) * 32 + 16 + nl] = f2bf(p1[r]);
    }
    asm volatile("s_wait_dscnt 0" ::: "memory");
    // re-read as A fragment: row = lane%16, K = half*8+0..7 / 16+half*8+0..7
    const unsigned short* pp = st + (size_t)nl * 32 + half * 8;
    union { uint4 qq[2]; v16bf v; } pf;
    pf.qq[0] = *(const uint4*)pp;
    pf.qq[1] = *(const uint4*)(pp + 16);

#pragma unroll
    for (int t = 0; t < 4; t++)
      o[t] = wmma_bf16(pf.v, load_b16(vh, NSEQ, t * 16, j0), o[t]);
  }

  int b_idx = bh >> 4, head = bh & 15;
#pragma unroll
  for (int t = 0; t < 4; t++) {
#pragma unroll
    for (int r = 0; r < 8; r++) {
      int i = i0 + half * 8 + r;
      int col = head * 64 + t * 16 + nl;
      ao[(size_t)(b_idx * NSEQ + i) * DIM_ + col] = f2bf(o[t][r] / lrow[r]);
    }
  }
}

// ---------------- output projection: [8192,1024] x [1024,1024]^T -> fp32 ----------------
__global__ __launch_bounds__(256) void proj_kernel(const unsigned short* __restrict__ ao,
                                                   const unsigned short* __restrict__ wo,
                                                   float* __restrict__ out) {
  int wid = blockIdx.x * 8 + (threadIdx.x >> 5);
  int mt = wid & 511, nt = wid >> 9;          // 512 m-tiles, 16 n-tiles
  int m0 = mt << 4, n0 = nt << 6;
  v8f acc[4] = {};
  for (int k0 = 0; k0 < DIM_; k0 += 32) {
    v16bf a = load_a16(ao, DIM_, m0, k0);
#pragma unroll
    for (int t = 0; t < 4; t++)
      acc[t] = wmma_bf16(a, load_b16(wo, DIM_, n0 + t * 16, k0), acc[t]);
  }
  int lane = threadIdx.x & 31, half = lane >> 4, nl = lane & 15;
#pragma unroll
  for (int t = 0; t < 4; t++)
#pragma unroll
    for (int r = 0; r < 8; r++)
      out[(size_t)(m0 + half * 8 + r) * DIM_ + n0 + t * 16 + nl] = acc[t][r];
}

extern "C" void kernel_launch(void* const* d_in, const int* in_sizes, int n_in,
                              void* d_out, int out_size, void* d_ws, size_t ws_size,
                              hipStream_t stream) {
  (void)in_sizes; (void)n_in; (void)out_size; (void)ws_size;
  const float* x     = (const float*)d_in[0];
  const float* ln_w  = (const float*)d_in[1];
  const float* ln_b  = (const float*)d_in[2];
  const float* w_qkv = (const float*)d_in[3];
  const float* w_out = (const float*)d_in[4];
  float* out = (float*)d_out;

  // bf16 workspace partition (~92 MB total), all segments 256B-aligned
  unsigned short* xn  = (unsigned short*)d_ws;                       // 8192x1024
  unsigned short* wq  = xn  + (size_t)ROWS * DIM_;                   // 3072x1024
  unsigned short* wo  = wq  + (size_t)3 * DIM_ * DIM_;               // 1024x1024
  unsigned short* qb  = wo  + (size_t)DIM_ * DIM_;                   // [b,h,n,64]
  unsigned short* kb  = qb  + (size_t)B_ * HEADS_ * NSEQ * DH_;      // [b,h,n,64]
  unsigned short* vtb = kb  + (size_t)B_ * HEADS_ * NSEQ * DH_;      // [b,h,64,n]
  unsigned short* ao  = vtb + (size_t)B_ * HEADS_ * NSEQ * DH_;      // 8192x1024

  cvt_kernel<<<1024, 256, 0, stream>>>(w_qkv, wq, 3 * DIM_ * DIM_);
  cvt_kernel<<<1024, 256, 0, stream>>>(w_out, wo, DIM_ * DIM_);
  ln_kernel<<<ROWS, 256, 0, stream>>>(x, ln_w, ln_b, xn);
  qkv_kernel<<<(512 * 48) / 8, 256, 0, stream>>>(xn, wq, qb, kb, vtb);
  flash_kernel<<<(64 * 128) / 4, 128, 0, stream>>>(qb, kb, vtb, ao);
  proj_kernel<<<(512 * 16) / 8, 256, 0, stream>>>(ao, wo, out);
}
